// AttentionRNN_20040317403333
// MI455X (gfx1250) — matrix operand order repaired and checked by reading, not compiled, not run on hardware
//
#include <hip/hip_runtime.h>
#include <stdint.h>

// ---------------------------------------------------------------------------
// AttentionRNN forward for MI455X (gfx1250, wave32, WMMA, TDM).
// V=32000, E=H=256, B=8, T=256.  Dominant cost: 2048x512 @ 512x32000 GEMM
// (67 GFLOP) -> bf16 WMMA with double-buffered TDM-staged LDS tiles.
// ---------------------------------------------------------------------------

#define DEV __device__ __forceinline__

typedef __attribute__((ext_vector_type(16))) __bf16        v16bf;
typedef __attribute__((ext_vector_type(8)))  float         v8f;
typedef __attribute__((ext_vector_type(4)))  unsigned int  v4u;
typedef __attribute__((ext_vector_type(8)))  int           v8i;
typedef __attribute__((ext_vector_type(4)))  int           v4i;
typedef unsigned short u16;

#ifndef HAVE_TDM
#if defined(__has_builtin)
#if __has_builtin(__builtin_amdgcn_tensor_load_to_lds)
#define HAVE_TDM 1
#endif
#endif
#endif
#ifndef HAVE_TDM
#define HAVE_TDM 0
#endif

static constexpr int Vv = 32000, Ee = 256, Hh = 256, Bb = 8, Tt = 256;
static constexpr int BT = Bb * Tt;          // 2048 rows
static constexpr int K2 = 2 * Hh;           // 512

// ---- fp32 -> bf16 (round-to-nearest-even), returns raw bits ---------------
DEV u16 f2bfb(float x) {
  unsigned u; __builtin_memcpy(&u, &x, 4);
  unsigned r = u + 0x7FFFu + ((u >> 16) & 1u);
  return (u16)(r >> 16);
}

union FragBF {
  v16bf v;
  u16 h[16];
  unsigned u[8];
};

// ---- WMMA fragment loaders -------------------------------------------------
// A-matrix 16x32 bf16: lane L holds row M=L&15; half=L>>4 selects
// K base 0/16; VGPR j -> K={2j,2j+1}+16*half (matches B fragment order).
DEV int a_frag_k(int j, int half) {
  return 2 * j + 16 * half;
}

// from global fp32, row-major [M x lda]
DEV v16bf load_a_frag_g(const float* A, int lda, int row0, int k0, int lane) {
  int r = lane & 15, half = lane >> 4;
  const float* base = A + (size_t)(row0 + r) * lda + k0;
  FragBF f;
#pragma unroll
  for (int j = 0; j < 8; ++j) {
    int kk = a_frag_k(j, half);
    f.h[2 * j]     = f2bfb(base[kk]);
    f.h[2 * j + 1] = f2bfb(base[kk + 1]);
  }
  return f.v;
}

// B-matrix 32x16 bf16 (K x N): lane holds col N=lane&15; lanes<16 -> K=0..15,
// lanes>=16 -> K=16..31, 2 K-values per VGPR.
DEV v16bf load_b_frag_g(const float* B, int ldb, int k0, int col0, int lane) {
  int c = col0 + (lane & 15), kb = k0 + (lane >> 4) * 16;
  FragBF f;
#pragma unroll
  for (int j = 0; j < 8; ++j) {
    f.h[2 * j]     = f2bfb(B[(size_t)(kb + 2 * j) * ldb + c]);
    f.h[2 * j + 1] = f2bfb(B[(size_t)(kb + 2 * j + 1) * ldb + c]);
  }
  return f.v;
}

// from LDS bf16 (single addrspace root so these lower to ds_load),
// base points at the 16x32 subtile origin.
DEV v16bf load_a_frag_lds(const u16* base, int ld, int lane) {
  int r = lane & 15, half = lane >> 4;
  FragBF f;
#pragma unroll
  for (int j = 0; j < 8; ++j) {
    int kk = a_frag_k(j, half);           // even -> dword aligned pair
    unsigned w; __builtin_memcpy(&w, base + r * ld + kk, 4);
    f.u[j] = w;
  }
  return f.v;
}

DEV v16bf load_b_frag_lds(const u16* base, int ld, int lane) {
  int c = lane & 15, kb = (lane >> 4) * 16;
  FragBF f;
#pragma unroll
  for (int j = 0; j < 8; ++j) {
    unsigned lo = base[(kb + 2 * j) * ld + c];
    unsigned hi = base[(kb + 2 * j + 1) * ld + c];
    f.u[j] = lo | (hi << 16);
  }
  return f.v;
}

// C/D 16x16 f32: VGPR i -> M = (lane>>4)*8 + i, N = lane&15.
DEV void store_c_tile(float* C, int ldc, int row0, int col0, v8f acc,
                      const float* bias, int lane) {
  int c = col0 + (lane & 15);
  int rbase = row0 + (lane >> 4) * 8;
  float bv = bias ? bias[c] : 0.0f;
#pragma unroll
  for (int i = 0; i < 8; ++i)
    C[(size_t)(rbase + i) * ldc + c] = acc[i] + bv;
}

// ---------------------------------------------------------------------------
// K0: embedding gather  emb[input[bt]] -> embWS[bt][:]
// ---------------------------------------------------------------------------
__global__ void gather_emb_kernel(const int* __restrict__ idx,
                                  const float* __restrict__ emb,
                                  float* __restrict__ out) {
  int bt = blockIdx.x;
  int i = threadIdx.x;
  out[(size_t)bt * Ee + i] = emb[(size_t)idx[bt] * Ee + i];
}

// ---------------------------------------------------------------------------
// K1/K3: generic  C[MxN] = A[MxK] @ B[KxN] + bias  via bf16 WMMA.
// One wave per 16x16 tile; K is compile-time so the loop fully unrolls into
// 8 back-to-back v_wmma with the f32->bf16 repack VALU co-issued (XDL WMMA
// is tracked as TRANS, so VALU co-executes per ISA 8.4).
// ---------------------------------------------------------------------------
template <int K>
__global__ void gemm_wmma_kernel(const float* __restrict__ A,
                                 const float* __restrict__ Bm,
                                 const float* __restrict__ bias,
                                 float* __restrict__ C,
                                 int M, int N) {
  int lane = threadIdx.x & 31;
  int w = blockIdx.x * (blockDim.x >> 5) + (threadIdx.x >> 5);
  int tilesN = N >> 4;
  int tm = w / tilesN, tn = w % tilesN;
  if (tm * 16 >= M) return;
  v8f acc = {0.f, 0.f, 0.f, 0.f, 0.f, 0.f, 0.f, 0.f};
#pragma unroll
  for (int k0 = 0; k0 < K; k0 += 32) {
    v16bf a = load_a_frag_g(A, K, tm * 16, k0, lane);
    v16bf b = load_b_frag_g(Bm, N, k0, tn * 16, lane);
    acc = __builtin_amdgcn_wmma_f32_16x16x32_bf16(
        false, a, false, b, (short)0, acc, false, false);
  }
  store_c_tile(C, N, tm * 16, tn * 16, acc, bias, lane);
}

// ---------------------------------------------------------------------------
// K2: RNN scan. One block per batch element (8 independent chains).
// h lives in LDS; W_hh (256 KB) is WGP$/L2 resident.
// ---------------------------------------------------------------------------
__global__ void rnn_scan_kernel(const float* __restrict__ xproj,
                                const float* __restrict__ W_hh,
                                const float* __restrict__ b_hh,
                                float* __restrict__ Hst) {
  __shared__ float hs[Hh];
  int b = blockIdx.x, i = threadIdx.x;
  hs[i] = 0.0f;
  __syncthreads();
  for (int t = 0; t < Tt; ++t) {
    float acc = xproj[((size_t)b * Tt + t) * Hh + i] + b_hh[i];
    if (t + 1 < Tt)   // hint next step's x_proj row toward the WGP caches
      __builtin_prefetch(&xproj[((size_t)b * Tt + t + 1) * Hh + i], 0, 1);
#pragma unroll 4
    for (int j = 0; j < Hh; ++j)
      acc = fmaf(hs[j], W_hh[j * Hh + i], acc);   // hs[j] = LDS broadcast
    float hn = tanhf(acc);
    __syncthreads();
    hs[i] = hn;
    Hst[((size_t)b * Tt + t) * Hh + i] = hn;
    __syncthreads();
  }
}

// ---------------------------------------------------------------------------
// K4: additive attention, one block per (b,t). 256 threads.
// scores_j = v . tanh(q_t + k_j), j < t; softmax; context = alpha @ Hst.
// ---------------------------------------------------------------------------
__global__ void attention_kernel(const float* __restrict__ q,
                                 const float* __restrict__ kmat,
                                 const float* __restrict__ vvec,
                                 const float* __restrict__ Hst,
                                 float* __restrict__ ctx) {
  __shared__ float qs[Hh];
  __shared__ float sc[Tt];
  __shared__ float red[Tt];
  int bt = blockIdx.x;
  int b = bt >> 8, t = bt & 255;
  int tid = threadIdx.x;

  qs[tid] = q[(size_t)bt * Hh + tid];
  __syncthreads();

  float s = -1e30f;
  if (tid < t) {
    const float* kr = kmat + ((size_t)(b << 8) + tid) * Hh;
    float a = 0.0f;
    for (int h = 0; h < Hh; ++h)
      a = fmaf(vvec[h], tanhf(qs[h] + kr[h]), a);
    s = a;
  }
  red[tid] = s;
  __syncthreads();
  for (int off = 128; off > 0; off >>= 1) {
    if (tid < off) red[tid] = fmaxf(red[tid], red[tid + off]);
    __syncthreads();
  }
  float mx = red[0];
  __syncthreads();
  float e = (tid < t) ? expf(s - mx) : 0.0f;
  sc[tid] = e;
  red[tid] = e;
  __syncthreads();
  for (int off = 128; off > 0; off >>= 1) {
    if (tid < off) red[tid] += red[tid + off];
    __syncthreads();
  }
  float inv = (t > 0) ? 1.0f / red[0] : 0.0f;

  // context[h=tid]; empty for t==0 -> zero (matches reference masking)
  float c = 0.0f;
  for (int j = 0; j < t; ++j)
    c = fmaf(sc[j], Hst[((size_t)(b << 8) + j) * Hh + tid], c);
  ctx[(size_t)bt * Hh + tid] = c * inv;
}

// ---------------------------------------------------------------------------
// K4.5: bf16 conversions for the big GEMM operands.
// ---------------------------------------------------------------------------
__global__ void build_comb16_kernel(const float* __restrict__ ctx,
                                    const float* __restrict__ Hst,
                                    u16* __restrict__ out) {
  size_t i = (size_t)blockIdx.x * blockDim.x + threadIdx.x;  // BT*K2 exact
  size_t row = i >> 9;
  int col = (int)(i & 511);
  float v = (col < Hh) ? ctx[row * Hh + col] : Hst[row * Hh + (col - Hh)];
  out[i] = f2bfb(v);
}

__global__ void cvt_bf16_kernel(const float* __restrict__ src,
                                u16* __restrict__ dst, size_t n) {
  for (size_t i = (size_t)blockIdx.x * blockDim.x + threadIdx.x; i < n;
       i += (size_t)gridDim.x * blockDim.x)
    dst[i] = f2bfb(src[i]);
}

// ---------------------------------------------------------------------------
// TDM: build D# (groups 0/1 per ISA 8.3/8.4, 2D tensor, data_size=2B) and
// issue tensor_load_to_lds (6-arg form on this toolchain:
// (v4u g0, v8i g1, v4i g2, v4i g3, v8i x, i32 cpol)). Groups 2/3 zero (2D).
// Dims/strides below are in elements (data_size units).
// ---------------------------------------------------------------------------
#if HAVE_TDM
DEV void tdm_load_2d(unsigned lds_off, const void* gptr,
                     unsigned tensor_d0, unsigned tensor_d1,
                     unsigned tile_d0, unsigned tile_d1,
                     unsigned long long stride0) {
  unsigned long long ga = (unsigned long long)(uintptr_t)gptr;
  v4u g0;
  g0.x = 1u;                                           // count=1, user mode
  g0.y = lds_off;                                      // lds_addr [63:32]
  g0.z = (unsigned)(ga & 0xFFFFFFFFu);                 // global_addr[31:0]
  g0.w = (unsigned)((ga >> 32) & 0x01FFFFFFu)          // global_addr[56:32]
         | (2u << 30);                                 // type = 2 (image)
  v8i g1;
  g1[0] = (int)(1u << 16);                             // data_size=1 -> 2B
  g1[1] = (int)((tensor_d0 & 0xFFFFu) << 16);          // tensor_dim0[15:0]
  g1[2] = (int)(((tensor_d0 >> 16) & 0xFFFFu) |
                ((tensor_d1 & 0xFFFFu) << 16));        // td0 hi | td1 lo
  g1[3] = (int)(((tensor_d1 >> 16) & 0xFFFFu) |
                ((tile_d0 & 0xFFFFu) << 16));          // td1 hi | tile_dim0
  g1[4] = (int)(tile_d1 & 0xFFFFu);                    // tile_dim1, tile_dim2=0
  g1[5] = (int)(unsigned)(stride0 & 0xFFFFFFFFull);    // dim0_stride[31:0]
  g1[6] = (int)(unsigned)((stride0 >> 32) & 0xFFFFull);// dim0_stride[47:32]
  g1[7] = 0;
  v4i gz4 = {0, 0, 0, 0};
  v8i gz8 = {0, 0, 0, 0, 0, 0, 0, 0};
  __builtin_amdgcn_tensor_load_to_lds(g0, g1, gz4, gz4, gz8, 0);
}
#endif

// ---------------------------------------------------------------------------
// K5: out = comb16[2048x512] @ w16[512x32000] + b_out.  Block tile 64x128,
// 8 waves x (32x32).  K staged 32-wide through double-buffered LDS via TDM:
// wave 0 issues the DMA for slice i+1 while all 8 waves run WMMA on slice i;
// s_wait_tensorcnt 0 + barrier publish the next buffer.
// LDS ping-pong: A@{0,12288} 64x32, B@{4096,16384} 32x128 (24 KB total).
// NOTE: all LDS accesses are smem + byte-offset off a single addrspace root
// so the backend emits ds_load (a pointer-array select defeats
// InferAddressSpaces and falls back to flat/global).
// ---------------------------------------------------------------------------
__global__ void out_gemm_kernel(const u16* __restrict__ A16,
                                const u16* __restrict__ W16,
                                const float* __restrict__ bias,
                                float* __restrict__ C) {
  const int K = K2, N = Vv, KSTEPS = K / 32;
  extern __shared__ char smem[];
  const int n0 = blockIdx.x * 128;
  const int m0 = blockIdx.y * 64;
  const int lane = threadIdx.x & 31;
  const int wave = threadIdx.x >> 5;
  const int wm = wave >> 2;           // 0..1 -> 32-row strip
  const int wn = wave & 3;            // 0..3 -> 32-col strip
  v8f acc[2][2];
#pragma unroll
  for (int a = 0; a < 2; ++a)
#pragma unroll
    for (int b = 0; b < 2; ++b)
#pragma unroll
      for (int i = 0; i < 8; ++i) acc[a][b][i] = 0.0f;

  int wave_s = __builtin_amdgcn_readfirstlane(wave);   // SGPR -> uniform branch

#if HAVE_TDM
  // ---- prologue: stage slice 0 into buffer 0 ----
  if (wave_s == 0) {
    tdm_load_2d(0u, A16 + (size_t)m0 * K,
                (unsigned)K, 64u, 32u, 64u, (unsigned long long)K);
    tdm_load_2d(4096u, W16 + n0,
                (unsigned)N, 32u, 128u, 32u, (unsigned long long)N);
    __builtin_amdgcn_s_wait_tensorcnt((short)0);
  }
  __syncthreads();
#pragma unroll 2
  for (int it = 0; it < KSTEPS; ++it) {
    unsigned curoff = (unsigned)(it & 1) * 12288u;
    unsigned nxtoff = 12288u - curoff;
    // prefetch next slice: the barrier ending iteration it-1 guarantees all
    // waves are done reading the nxt buffer, so it is safe to overwrite now.
    if (wave_s == 0 && it + 1 < KSTEPS) {
      int k1 = (it + 1) * 32;
      tdm_load_2d(nxtoff, A16 + (size_t)m0 * K + k1,
                  (unsigned)K, 64u, 32u, 64u, (unsigned long long)K);
      tdm_load_2d(nxtoff + 4096u, W16 + (size_t)k1 * N + n0,
                  (unsigned)N, 32u, 128u, 32u, (unsigned long long)N);
    }
    // compute on current slice (overlaps the in-flight TDM)
    const u16* Asc = (const u16*)(smem + curoff);            // 64 x 32
    const u16* Bsc = (const u16*)(smem + curoff + 4096u);    // 32 x 128
#pragma unroll
    for (int ti = 0; ti < 2; ++ti) {
      v16bf a = load_a_frag_lds(Asc + (wm * 32 + ti * 16) * 32, 32, lane);
#pragma unroll
      for (int tj = 0; tj < 2; ++tj) {
        v16bf b = load_b_frag_lds(Bsc + wn * 32 + tj * 16, 128, lane);
        acc[ti][tj] = __builtin_amdgcn_wmma_f32_16x16x32_bf16(
            false, a, false, b, (short)0, acc[ti][tj], false, false);
      }
    }
    if (wave_s == 0 && it + 1 < KSTEPS)
      __builtin_amdgcn_s_wait_tensorcnt((short)0);
    __syncthreads();                 // publish nxt buffer / retire cur reads
  }
#else
  u16* As0 = (u16*)smem;
  u16* Bs0 = (u16*)(smem + 4096);
  for (int it = 0; it < KSTEPS; ++it) {
    int k0 = it * 32;
    __syncthreads();
    for (int i = threadIdx.x; i < 64 * 32; i += 256) {
      int r = i >> 5, cc = i & 31;
      As0[i] = A16[(size_t)(m0 + r) * K + k0 + cc];
    }
    for (int i = threadIdx.x; i < 32 * 128; i += 256) {
      int r = i >> 7, cc = i & 127;
      Bs0[i] = W16[(size_t)(k0 + r) * N + n0 + cc];
    }
    __syncthreads();
#pragma unroll
    for (int ti = 0; ti < 2; ++ti) {
      v16bf a = load_a_frag_lds(As0 + (wm * 32 + ti * 16) * 32, 32, lane);
#pragma unroll
      for (int tj = 0; tj < 2; ++tj) {
        v16bf b = load_b_frag_lds(Bs0 + wn * 32 + tj * 16, 128, lane);
        acc[ti][tj] = __builtin_amdgcn_wmma_f32_16x16x32_bf16(
            false, a, false, b, (short)0, acc[ti][tj], false, false);
      }
    }
  }
#endif

#pragma unroll
  for (int ti = 0; ti < 2; ++ti)
#pragma unroll
    for (int tj = 0; tj < 2; ++tj)
      store_c_tile(C, N, m0 + wm * 32 + ti * 16, n0 + wn * 32 + tj * 16,
                   acc[ti][tj], bias, lane);
}

// ---------------------------------------------------------------------------
// Host launcher
// ---------------------------------------------------------------------------
extern "C" void kernel_launch(void* const* d_in, const int* in_sizes, int n_in,
                              void* d_out, int out_size, void* d_ws,
                              size_t ws_size, hipStream_t stream) {
  (void)in_sizes; (void)n_in; (void)out_size; (void)ws_size;
  const int*   input     = (const int*)d_in[0];
  const float* embedding = (const float*)d_in[1];
  const float* W_ih      = (const float*)d_in[2];
  const float* b_ih      = (const float*)d_in[3];
  const float* W_hh      = (const float*)d_in[4];
  const float* b_hh      = (const float*)d_in[5];
  const float* W_ac      = (const float*)d_in[6];
  const float* b_ac      = (const float*)d_in[7];
  const float* W_ap      = (const float*)d_in[8];
  const float* b_ap      = (const float*)d_in[9];
  const float* v_attn    = (const float*)d_in[10];
  const float* W_out     = (const float*)d_in[11];
  const float* b_out     = (const float*)d_in[12];
  float* out = (float*)d_out;

  // workspace layout (~46 MB)
  float* embWS = (float*)d_ws;                 // 2048*256 f32
  float* xproj = embWS + (size_t)BT * Hh;      // 2048*256
  float* Hst   = xproj + (size_t)BT * Hh;      // 2048*256
  float* qb    = Hst + (size_t)BT * Hh;        // 2048*256
  float* kb    = qb + (size_t)BT * Hh;         // 2048*256
  float* ctx   = kb + (size_t)BT * Hh;         // 2048*256
  u16* comb16  = (u16*)(ctx + (size_t)BT * Hh);        // 2048*512 bf16
  u16* w16     = comb16 + (size_t)BT * K2;             // 512*32000 bf16

  gather_emb_kernel<<<BT, Hh, 0, stream>>>(input, embedding, embWS);
  // x_proj = emb @ W_ih + b_ih : 2048 tiles -> 256 blocks * 8 waves
  gemm_wmma_kernel<Ee><<<256, 256, 0, stream>>>(embWS, W_ih, b_ih, xproj,
                                                BT, Hh);
  rnn_scan_kernel<<<Bb, Hh, 0, stream>>>(xproj, W_hh, b_hh, Hst);
  gemm_wmma_kernel<Hh><<<256, 256, 0, stream>>>(Hst, W_ac, b_ac, qb, BT, Hh);
  gemm_wmma_kernel<Hh><<<256, 256, 0, stream>>>(Hst, W_ap, b_ap, kb, BT, Hh);
  attention_kernel<<<BT, Hh, 0, stream>>>(qb, kb, v_attn, Hst, ctx);
  build_comb16_kernel<<<(BT * K2) / 256, 256, 0, stream>>>(ctx, Hst, comb16);
  cvt_bf16_kernel<<<4096, 256, 0, stream>>>(W_out, w16, (size_t)K2 * Vv);
  // out = comb16 @ w16 + b_out : grid 250 x 32, 24 KB ping-pong LDS
  out_gemm_kernel<<<dim3(Vv / 128, BT / 64), 256, 24576, stream>>>(
      comb16, w16, b_out, out);
}
